// BinaryMemoryRNN_56873956934276
// MI455X (gfx1250) — matrix operand order repaired and checked by reading, not compiled, not run on hardware
//
#include <hip/hip_runtime.h>
#include <hip/hip_bf16.h>

// ---------------- problem constants (match reference) ----------------
#define BB   8192      // batch
#define II   1024      // input features
#define HH   1024      // hidden
#define NBB  14        // bits
#define NSTEP 128      // K steps: 4096 / 32
#define EPSL 1e-5f

typedef __attribute__((ext_vector_type(16))) __bf16 v16bf;
typedef __attribute__((ext_vector_type(8)))  float  v8f;
typedef __attribute__((ext_vector_type(4)))  float  vf4;   // plain vector (no HIP class)

// explicit global-address-space pointers: force global_load_b128 (LOADcnt
// only) instead of flat_load_b128 (which also ties up DScnt and would make
// the WMMA fragment s_wait_dscnt fences wait on staged global loads).
// All conversions go through integers (always-legal cast path).
typedef __attribute__((address_space(1))) const float gfloat;
typedef __attribute__((address_space(1))) const vf4   gvf4;

union FragBF { uint4 u[2]; v16bf v; };

// pack two f32 -> packed bf16x2 (a -> low half), single HW instruction:
// V_CVT_PK_BF16_F32: D[15:0]=bf16(S0), D[31:16]=bf16(S1), round-nearest-even
__device__ __forceinline__ unsigned int pk2bf(float a, float b) {
    unsigned int r;
    asm("v_cvt_pk_bf16_f32 %0, %1, %2" : "=v"(r) : "v"(a), "v"(b));
    return r;
}

// =====================================================================
// Kernel 1: logits = h_prev @ Mw.T + Mb  ->  binary codes -> indices
// One wave32 per batch row. 4 waves per 128-thread block.
// =====================================================================
__global__ void k_logits_idx(const float* __restrict__ h_prev,
                             const float* __restrict__ Mw,
                             const float* __restrict__ Mb,
                             int* __restrict__ idx1,
                             int* __restrict__ idx2) {
    const int lane = threadIdx.x & 31;
    const int wave = threadIdx.x >> 5;
    const int row  = blockIdx.x * 4 + wave;
    if (row >= BB) return;

    float acc[2 * NBB];
#pragma unroll
    for (int j = 0; j < 2 * NBB; ++j) acc[j] = 0.f;

    const float* hrow = h_prev + (size_t)row * HH;
    for (int k = lane; k < HH; k += 32) {
        float h = hrow[k];
#pragma unroll
        for (int j = 0; j < 2 * NBB; ++j)
            acc[j] = fmaf(h, Mw[(size_t)j * HH + k], acc[j]);
    }
#pragma unroll
    for (int j = 0; j < 2 * NBB; ++j) {
#pragma unroll
        for (int off = 16; off > 0; off >>= 1)
            acc[j] += __shfl_xor(acc[j], off, 32);
    }
    if (lane == 0) {
        int s1 = 0, s2 = 0;
#pragma unroll
        for (int j = 0; j < NBB; ++j) {
            if (acc[j]       + Mb[j]       > 0.f) s1 |= 1 << (NBB - 1 - j);
            if (acc[NBB + j] + Mb[NBB + j] > 0.f) s2 |= 1 << (NBB - 1 - j);
        }
        const int split = (1 << NBB) / 2;              // 8192
        idx1[row] = (s1 < split) ? s1 : (split - 1);   // clip [0, 8191]
        idx2[row] = (s2 < split) ? split : s2;         // clip [8192, 16383]
    }
}

// =====================================================================
// Kernel 2: fused 4-way GEMM as one [8192 x 1024 x 4096] bf16 WMMA GEMM
//   pre[b,n] = sum_seg A_seg[b,:] . W_seg[n,:]  + (Wb+Ub+Qrb+Qlb)[n]
// Tile: 128(M) x 128(N) per 256-thread workgroup. K-step 32.
// Waves in a 4x2 grid: each wave owns 32(M) x 64(N) -> 2x4 C fragments.
// Double-buffered LDS: one barrier per K-step, global loads for step s+1
// overlap with the 8 WMMAs of step s.
// =====================================================================
__global__ void __launch_bounds__(256)
k_fused_gemm(const float* __restrict__ x,
             const float* __restrict__ h_prev,
             const float* __restrict__ memory,
             const float* __restrict__ Ww, const float* __restrict__ Uw,
             const float* __restrict__ Qrw, const float* __restrict__ Qlw,
             const float* __restrict__ Wb, const float* __restrict__ Ub,
             const float* __restrict__ Qrb, const float* __restrict__ Qlb,
             const int* __restrict__ idx1, const int* __restrict__ idx2,
             float* __restrict__ pre) {
    // ping-pong LDS tiles; 40-ushort (80B) pitch -> conflict-free 16B reads
    __shared__ unsigned short As[2][128][40];   // A tile: [m][k] bf16
    __shared__ unsigned short Bs[2][128][40];   // B tile: [n][k] bf16 (W rows)
    __shared__ int sI1[128], sI2[128];

    const int tid   = threadIdx.x;
    const int wave  = tid >> 5;
    const int lane  = tid & 31;
    const int g     = lane >> 4;       // lane group (0/1)
    const int ln    = lane & 15;
    const int wm    = wave >> 1;       // wave row   (0..3) -> 32 M rows each
    const int wn    = wave & 1;        // wave col   (0..1) -> 64 N cols each
    const int tileN = blockIdx.x * 128;
    const int tileM = blockIdx.y * 128;

    // cache gather indices for this row block
    if (tid < 128)       sI1[tid]       = idx1[tileM + tid];
    else                 sI2[tid - 128] = idx2[tileM + tid - 128];
    __syncthreads();

    // loader role: thread covers row r, k-half hs (16 floats per K-step)
    const int r  = tid >> 1;
    const int hs = (tid & 1) * 16;
    const int rg = tileM + r;            // global A row
    const int ng = tileN + r;            // global W row (= output col)

    // per-segment source base addresses (as integers; cast to AS1 at use)
    const unsigned long long abase[4] = {
        (unsigned long long)(x      + (size_t)rg * II      + hs),
        (unsigned long long)(h_prev + (size_t)rg * HH      + hs),
        (unsigned long long)(memory + (size_t)sI1[r] * HH  + hs),
        (unsigned long long)(memory + (size_t)sI2[r] * HH  + hs) };
    const unsigned long long wbase[4] = {
        (unsigned long long)(Ww  + (size_t)ng * HH + hs),
        (unsigned long long)(Uw  + (size_t)ng * HH + hs),
        (unsigned long long)(Qrw + (size_t)ng * HH + hs),
        (unsigned long long)(Qlw + (size_t)ng * HH + hs) };

    v8f acc[2][4];
#pragma unroll
    for (int mi = 0; mi < 2; ++mi)
#pragma unroll
        for (int ni = 0; ni < 4; ++ni)
#pragma unroll
            for (int e = 0; e < 8; ++e) acc[mi][ni][e] = 0.f;

    // prologue: stage step 0 into registers
    vf4 va[4], vb[4];
    {
        const gfloat* ap = (const gfloat*)abase[0];
        const gfloat* wp = (const gfloat*)wbase[0];
#pragma unroll
        for (int q = 0; q < 4; ++q) {
            va[q] = *(const gvf4*)(ap + q * 4);
            vb[q] = *(const gvf4*)(wp + q * 4);
        }
    }

    for (int s = 0; s < NSTEP; ++s) {
        const int p = s & 1;

        // ---- convert staged registers -> bf16 LDS (buffer p) ----
#pragma unroll
        for (int q = 0; q < 4; ++q) {
            const int k0 = hs + q * 4;
            *(unsigned int*)&As[p][r][k0]     = pk2bf(va[q].x, va[q].y);
            *(unsigned int*)&As[p][r][k0 + 2] = pk2bf(va[q].z, va[q].w);
            *(unsigned int*)&Bs[p][r][k0]     = pk2bf(vb[q].x, vb[q].y);
            *(unsigned int*)&Bs[p][r][k0 + 2] = pk2bf(vb[q].z, vb[q].w);
        }
        __syncthreads();   // single barrier per K-step

        // ---- issue global loads for step s+1 (hide under WMMAs) ----
        if (s + 1 < NSTEP) {
            const int sn = s + 1;
            const gfloat* ap = (const gfloat*)abase[sn >> 5] + (size_t)(sn & 31) * 32;
            const gfloat* wp = (const gfloat*)wbase[sn >> 5] + (size_t)(sn & 31) * 32;
#pragma unroll
            for (int q = 0; q < 4; ++q) {
                va[q] = *(const gvf4*)(ap + q * 4);
                vb[q] = *(const gvf4*)(wp + q * 4);
            }
            if (s + 2 < NSTEP) {   // L2 prefetch two steps ahead
                const int sp = s + 2;
                __builtin_prefetch(
                    (const void*)(abase[sp >> 5] + (unsigned long long)(sp & 31) * 128), 0, 1);
                __builtin_prefetch(
                    (const void*)(wbase[sp >> 5] + (unsigned long long)(sp & 31) * 128), 0, 1);
            }
        }

        // ---- compute: 2 A-frags, 4 B-frags, 8 WMMAs ----
        // A frag (16x32): lane ln = M row, elems k = (i/8)*16 + g*8 + (i%8)
        FragBF a0, a1;
        a0.u[0] = *(const uint4*)&As[p][wm * 32 + ln][g * 8];
        a0.u[1] = *(const uint4*)&As[p][wm * 32 + ln][16 + g * 8];
        a1.u[0] = *(const uint4*)&As[p][wm * 32 + 16 + ln][g * 8];
        a1.u[1] = *(const uint4*)&As[p][wm * 32 + 16 + ln][16 + g * 8];
#pragma unroll
        for (int ni = 0; ni < 4; ++ni) {
            // B frag (32x16): lane ln = N col, elems k = (i/8)*16 + g*8 + (i%8)
            FragBF b;
            b.u[0] = *(const uint4*)&Bs[p][wn * 64 + ni * 16 + ln][g * 8];
            b.u[1] = *(const uint4*)&Bs[p][wn * 64 + ni * 16 + ln][16 + g * 8];
            acc[0][ni] = __builtin_amdgcn_wmma_f32_16x16x32_bf16(
                false, a0.v, false, b.v, (short)0, acc[0][ni], false, false);
            acc[1][ni] = __builtin_amdgcn_wmma_f32_16x16x32_bf16(
                false, a1.v, false, b.v, (short)0, acc[1][ni], false, false);
        }
    }

    // ---- epilogue: add fused bias, store pre (fp32) ----
#pragma unroll
    for (int ni = 0; ni < 4; ++ni) {
        const int col  = tileN + wn * 64 + ni * 16 + ln;
        const float bs = Wb[col] + Ub[col] + Qrb[col] + Qlb[col];
#pragma unroll
        for (int mi = 0; mi < 2; ++mi)
#pragma unroll
            for (int j = 0; j < 8; ++j) {
                // C layout: VGPR j holds M = 8*g + j
                const int rowg = tileM + wm * 32 + mi * 16 + g * 8 + j;
                pre[(size_t)rowg * HH + col] = acc[mi][ni][j] + bs;
            }
    }
}

// =====================================================================
// Kernel 3: row LayerNorm + sigmoid. One 256-thread block per row.
// =====================================================================
__global__ void __launch_bounds__(256)
k_ln_sigmoid(const float* __restrict__ pre,
             const float* __restrict__ gamma,
             const float* __restrict__ beta,
             float* __restrict__ out) {
    __shared__ float ss[8], sq[8];
    const int row  = blockIdx.x;
    const int tid  = threadIdx.x;
    const int lane = tid & 31;
    const int wave = tid >> 5;
    const int c    = tid * 4;

    const float4 v = *(const float4*)(pre + (size_t)row * HH + c);
    float s = v.x + v.y + v.z + v.w;
    float q = v.x * v.x + v.y * v.y + v.z * v.z + v.w * v.w;
#pragma unroll
    for (int off = 16; off > 0; off >>= 1) {
        s += __shfl_xor(s, off, 32);
        q += __shfl_xor(q, off, 32);
    }
    if (lane == 0) { ss[wave] = s; sq[wave] = q; }
    __syncthreads();
    float ts = 0.f, tq = 0.f;
#pragma unroll
    for (int i = 0; i < 8; ++i) { ts += ss[i]; tq += sq[i]; }
    const float mu  = ts * (1.f / HH);
    const float var = tq * (1.f / HH) - mu * mu;
    const float rs  = rsqrtf(var + EPSL);

    const float4 gm = *(const float4*)(gamma + c);
    const float4 bt = *(const float4*)(beta + c);
    float4 o;
    o.x = 1.f / (1.f + __expf(-((v.x - mu) * rs * gm.x + bt.x)));
    o.y = 1.f / (1.f + __expf(-((v.y - mu) * rs * gm.y + bt.y)));
    o.z = 1.f / (1.f + __expf(-((v.z - mu) * rs * gm.z + bt.z)));
    o.w = 1.f / (1.f + __expf(-((v.w - mu) * rs * gm.w + bt.w)));
    *(float4*)(out + (size_t)row * HH + c) = o;
}

// =====================================================================
extern "C" void kernel_launch(void* const* d_in, const int* in_sizes, int n_in,
                              void* d_out, int out_size, void* d_ws, size_t ws_size,
                              hipStream_t stream) {
    const float* x      = (const float*)d_in[0];
    const float* h_prev = (const float*)d_in[1];
    const float* memory = (const float*)d_in[2];
    const float* Ww     = (const float*)d_in[3];
    const float* Wb     = (const float*)d_in[4];
    const float* Uw     = (const float*)d_in[5];
    const float* Ub     = (const float*)d_in[6];
    const float* Mw     = (const float*)d_in[7];
    const float* Mb     = (const float*)d_in[8];
    const float* Qrw    = (const float*)d_in[9];
    const float* Qrb    = (const float*)d_in[10];
    const float* Qlw    = (const float*)d_in[11];
    const float* Qlb    = (const float*)d_in[12];
    const float* gamma  = (const float*)d_in[13];
    const float* beta   = (const float*)d_in[14];

    // workspace layout: pre [B*H fp32] | idx1 [B i32] | idx2 [B i32]
    float* pre  = (float*)d_ws;
    int*   idx1 = (int*)((char*)d_ws + (size_t)BB * HH * sizeof(float));
    int*   idx2 = idx1 + BB;

    k_logits_idx<<<dim3(BB / 4), dim3(128), 0, stream>>>(h_prev, Mw, Mb, idx1, idx2);

    k_fused_gemm<<<dim3(HH / 128, BB / 128), dim3(256), 0, stream>>>(
        x, h_prev, memory, Ww, Uw, Qrw, Qlw, Wb, Ub, Qrb, Qlb, idx1, idx2, pre);

    k_ln_sigmoid<<<dim3(BB), dim3(256), 0, stream>>>(pre, gamma, beta, (float*)d_out);
}
